// SSD_78305843740943
// MI455X (gfx1250) — matrix operand order repaired and checked by reading, not compiled, not run on hardware
//
#include <hip/hip_runtime.h>
#include <math.h>
#include <stdint.h>

#define BATCH 4
#define IMG 300
#define NCLS 81
#define CCLS 80
#define THRS 0.02f
#define IOU_THR 0.45f
#define MAXDET 200
#define ANCH_TOT 8732
#define KTOT 5320
#define SORTN 8192

typedef unsigned short ushort_t;
typedef __attribute__((ext_vector_type(16))) __bf16 v16bf;
typedef __attribute__((ext_vector_type(8)))  float  v8f;

__device__ __forceinline__ ushort_t f2bf(float f) {
  unsigned int u = __float_as_uint(f);
  u += 0x7fffu + ((u >> 16) & 1u);          // round-to-nearest-even
  return (ushort_t)(u >> 16);
}
__device__ __forceinline__ float bf2f(ushort_t h) {
  return __uint_as_float(((unsigned int)h) << 16);
}

union BF16Frag { ushort_t u[16]; unsigned int w[8]; v16bf v; };

// ---- CDNA5 async global->LDS copy (ASYNCcnt tracked, CDNA5 ISA) --------
__device__ __forceinline__ void async_g2l_b128(unsigned lds_byte_addr, const void* gaddr) {
  asm volatile("global_load_async_to_lds_b128 %0, %1, off"
               :: "v"(lds_byte_addr), "v"((unsigned long long)(uintptr_t)gaddr)
               : "memory");
}
__device__ __forceinline__ void wait_async0() {
#if __has_builtin(__builtin_amdgcn_s_wait_asynccnt)
  __builtin_amdgcn_s_wait_asynccnt(0);
#else
  asm volatile("s_wait_asynccnt 0x0" ::: "memory");
#endif
}

// ---------------------------------------------------------------- conversions
__global__ void cvt_f32_bf16(const float* __restrict__ in, ushort_t* __restrict__ out, int n) {
  int i = blockIdx.x * 256 + threadIdx.x;
  if (i < n) out[i] = f2bf(in[i]);
}

// repack fp32 OIHW weights -> zero-padded bf16 [Mpad x Kpad] GEMM A matrix
__global__ void repack_w(const float* __restrict__ w, ushort_t* __restrict__ out,
                         int Kdim, int Kpad, int Cout, int Mpad) {
  int idx = blockIdx.x * 256 + threadIdx.x;
  int tot = Mpad * Kpad;
  if (idx >= tot) return;
  int row = idx / Kpad, k = idx - row * Kpad;
  float v = (row < Cout && k < Kdim) ? w[(size_t)row * Kdim + k] : 0.f;
  out[idx] = f2bf(v);
}

// ------------------------------------------------- implicit-GEMM conv (WMMA)
// M = Cout (tile 128), N = BATCH*Hout*Wout (tile 64), K stepped by 32.
// 128 threads = 4 waves; each wave owns 2 A-fragments (32 M rows) and reuses
// each B-fragment across both -> 8 WMMAs per K-step per wave.
// A tile staged via global_load_async_to_lds_b128 (weights pre-padded).
__global__ __launch_bounds__(128) void conv_bf16_wmma(
    const ushort_t* __restrict__ in, const ushort_t* __restrict__ wgt,
    const float* __restrict__ bias,
    ushort_t* __restrict__ out_bf, float* __restrict__ out_f32,
    int Cin, int Hin, int Win, int Cout, int Hout, int Wout,
    int Kk, int stride, int pad, int dil, int relu, int Kpad)
{
  __shared__ ushort_t As[128 * 32];   // [m][k]  8 KB
  __shared__ ushort_t Bs[64 * 32];    // [n][k]  4 KB (transposed)
  const int Kdim = Cin * Kk * Kk;
  const int KK2  = Kk * Kk;
  const int HWo  = Hout * Wout;
  const int Ntot = BATCH * HWo;
  const int m0   = blockIdx.y * 128;
  const int n0   = blockIdx.x * 64;
  const int tid  = threadIdx.x;
  const int lane = tid & 31;
  const int wv   = tid >> 5;

  // ---- hoisted per-thread B-staging geometry (n fixed per thread) ----
  const int nn   = tid & 63;          // column of Bs this thread fills
  const int kfix = tid >> 6;          // 0/1: k parity class
  int n = n0 + nn;
  bool nvalid = n < Ntot;
  int b = 0, ho = 0, wo = 0;
  if (nvalid) { b = n / HWo; int r = n - b * HWo; ho = r / Wout; wo = r - ho * Wout; }
  const int hbase = ho * stride - pad;
  const int wbase = wo * stride - pad;
  const ushort_t* inb = in + (size_t)b * Cin * Hin * Win;

  v8f zero = {0.f, 0.f, 0.f, 0.f, 0.f, 0.f, 0.f, 0.f};
  v8f acc[2][4] = {{zero, zero, zero, zero}, {zero, zero, zero, zero}};

  const unsigned asBase = (unsigned)(uintptr_t)&As[0];

  for (int k0 = 0; k0 < Kpad; k0 += 32) {
    // ---- A tile: 128x32 bf16 = 8 KB, async global->LDS in 16B chunks ----
    const ushort_t* wrow = wgt + (size_t)m0 * Kpad + k0;
    #pragma unroll
    for (int p = 0; p < 4; ++p) {
      int idx = tid + (p << 7);              // 0..511 chunk id
      int row = idx >> 2, q = idx & 3;       // 4x16B chunks per 64B row
      async_g2l_b128(asBase + (unsigned)((row << 6) + (q << 4)),
                     wrow + (size_t)row * Kpad + (q << 3));
    }
    // ---- B tile: im2col gather, n-major (coalesced along wo) ----
    #pragma unroll
    for (int p = 0; p < 16; ++p) {
      int kk = kfix + (p << 1);
      int kg = k0 + kk;
      ushort_t val = 0;
      if (nvalid && kg < Kdim) {
        int ci = kg / KK2;
        int rr = kg - ci * KK2;
        int kh = rr / Kk, kw = rr - kh * Kk;
        int hi = hbase + kh * dil;
        int wi = wbase + kw * dil;
        if (hi >= 0 && hi < Hin && wi >= 0 && wi < Win)
          val = inb[((size_t)ci * Hin + hi) * Win + wi];
      }
      Bs[nn * 32 + kk] = val;
    }
    wait_async0();
    __syncthreads();

    // ---- fragments + 8 WMMAs ----
    BF16Frag Af[2];
    {
      int kb = (lane < 16) ? 0 : 8;
      #pragma unroll
      for (int f = 0; f < 2; ++f) {
        int am = (wv << 5) + (f << 4) + (lane & 15);
        #pragma unroll
        for (int j = 0; j < 8; ++j) {
          int k = kb + ((j & 4) << 2) + ((j & 3) << 1);
          Af[f].w[j] = *(const unsigned int*)&As[am * 32 + k];
        }
      }
    }
    int kbb = (lane < 16) ? 0 : 8;
    #pragma unroll
    for (int nb = 0; nb < 4; ++nb) {
      BF16Frag Bf;
      int bn = (nb << 4) + (lane & 15);
      #pragma unroll
      for (int j = 0; j < 8; ++j)
        Bf.w[j] = *(const unsigned int*)&Bs[bn * 32 + kbb + ((j & 4) << 2) + ((j & 3) << 1)];
      acc[0][nb] = __builtin_amdgcn_wmma_f32_16x16x32_bf16(
          false, Af[0].v, false, Bf.v, (short)0, acc[0][nb], false, false);
      acc[1][nb] = __builtin_amdgcn_wmma_f32_16x16x32_bf16(
          false, Af[1].v, false, Bf.v, (short)0, acc[1][nb], false, false);
    }
    __syncthreads();
  }

  // ---- epilogue: C/D layout -> lane n = lane%16, VGPR r -> M = r+8*(lane>=16)
  #pragma unroll
  for (int nb = 0; nb < 4; ++nb) {
    int nOut = n0 + (nb << 4) + (lane & 15);
    if (nOut >= Ntot) continue;
    int ob = nOut / HWo;
    int orr = nOut - ob * HWo;
    int oho = orr / Wout, owo = orr - oho * Wout;
    #pragma unroll
    for (int f = 0; f < 2; ++f) {
      int mb = m0 + (wv << 5) + (f << 4) + ((lane < 16) ? 0 : 8);
      #pragma unroll
      for (int ri = 0; ri < 8; ++ri) {
        int m = mb + ri;
        if (m >= Cout) continue;
        float vv = acc[f][nb][ri] + bias[m];
        if (relu) vv = fmaxf(vv, 0.f);
        size_t o = ((size_t)(ob * Cout + m) * Hout + oho) * Wout + owo;
        if (out_f32) out_f32[o] = vv;
        if (out_bf)  out_bf[o]  = f2bf(vv);
      }
    }
  }
}

// ----------------------------------------------------------------- maxpool
__global__ void maxpool_bf16(const ushort_t* __restrict__ in, ushort_t* __restrict__ out,
                             int C_, int Hin, int Win, int Hout, int Wout,
                             int k, int s, int p) {
  int idx = blockIdx.x * 256 + threadIdx.x;
  int tot = BATCH * C_ * Hout * Wout;
  if (idx >= tot) return;
  int wo = idx % Wout; int t = idx / Wout;
  int ho = t % Hout;  t /= Hout;
  int c  = t % C_;    int b = t / C_;
  float m = -3.0e38f;
  for (int i = 0; i < k; ++i) {
    int hi = ho * s - p + i;
    if (hi < 0 || hi >= Hin) continue;
    for (int j = 0; j < k; ++j) {
      int wi = wo * s - p + j;
      if (wi < 0 || wi >= Win) continue;
      m = fmaxf(m, bf2f(in[((size_t)(b * C_ + c) * Hin + hi) * Win + wi]));
    }
  }
  out[idx] = f2bf(m);
}

// ----------------------------------------------------------------- L2 norm
__global__ void l2norm_bf16(const ushort_t* __restrict__ in, const float* __restrict__ scale,
                            ushort_t* __restrict__ out, int C_, int HW) {
  int idx = blockIdx.x * 256 + threadIdx.x;
  if (idx >= BATCH * HW) return;
  int b = idx / HW, p = idx % HW;
  const ushort_t* base = in + (size_t)b * C_ * HW + p;
  float sum = 0.f;
  for (int c = 0; c < C_; ++c) { float v = bf2f(base[(size_t)c * HW]); sum += v * v; }
  float inv = 1.0f / fmaxf(sqrtf(sum), 1e-12f);
  ushort_t* ob = out + (size_t)b * C_ * HW + p;
  for (int c = 0; c < C_; ++c)
    ob[(size_t)c * HW] = f2bf(bf2f(base[(size_t)c * HW]) * inv * scale[c]);
}

// ------------------------------------------------------- head permute+softmax
__global__ void head_softmax(const float* __restrict__ y, float* __restrict__ scr,
                             int A, int H, int W, int aoff) {
  int HWA = H * W * A;
  int idx = blockIdx.x * 256 + threadIdx.x;
  if (idx >= BATCH * HWA) return;
  int b = idx / HWA, pos = idx % HWA;
  int a = pos % A; int pw = pos / A;
  int h = pw / W,  w = pw % W;
  int HW = H * W;
  const float* base = y + (((size_t)(b * A * NCLS + a * NCLS)) * H + h) * W + w;
  float mx = -3.0e38f;
  for (int c = 0; c < NCLS; ++c) mx = fmaxf(mx, base[(size_t)c * HW]);
  float sum = 0.f;
  for (int c = 0; c < NCLS; ++c) sum += __expf(base[(size_t)c * HW] - mx);
  float inv = 1.0f / sum;
  float* o = scr + ((size_t)b * ANCH_TOT + aoff + pos) * CCLS;
  for (int c = 0; c < CCLS; ++c) o[c] = __expf(base[(size_t)c * HW] - mx) * inv;
}

__global__ void head_reg(const float* __restrict__ y, float* __restrict__ regf,
                         int A, int H, int W, int aoff) {
  int HWA = H * W * A;
  int idx = blockIdx.x * 256 + threadIdx.x;
  if (idx >= BATCH * HWA) return;
  int b = idx / HWA, pos = idx % HWA;
  int a = pos % A; int pw = pos / A;
  int h = pw / W,  w = pw % W;
  int HW = H * W;
  float* o = regf + ((size_t)b * ANCH_TOT + aoff + pos) * 4;
  for (int j = 0; j < 4; ++j)
    o[j] = y[(((size_t)(b * A * 4 + a * 4 + j)) * H + h) * W + w];
}

// ----------------------------------------------------------------- priors
__global__ void build_priors(float* __restrict__ priors, int aoff, int f, int A,
                             float st, float bs, float bn) {
  int idx = blockIdx.x * 256 + threadIdx.x;
  if (idx >= f * f * A) return;
  int a = idx % A; int cell = idx / A;
  int row = cell / f, col = cell % f;
  float x = (col + 0.5f) * st, y = (row + 0.5f) * st;
  const float s2 = 1.41421356237f, s3 = 1.73205080757f;
  float w, h;
  switch (a) {
    case 0:  w = bs; h = bs; break;
    case 1:  { float e = sqrtf(bn * bs); w = e; h = e; } break;
    case 2:  w = bs * s2; h = bs / s2; break;
    case 3:  w = bs / s2; h = bs * s2; break;
    case 4:  w = bs * s3; h = bs / s3; break;
    default: w = bs / s3; h = bs * s3; break;
  }
  float* o = priors + (size_t)(aoff + idx) * 4;
  o[0] = x; o[1] = y; o[2] = w; o[3] = h;
}

// ----------------------------------------------- per-level deterministic top-k
__global__ __launch_bounds__(256) void topk_level(
    const float* __restrict__ scr, int lf, int kcap, int aoff, int koff,
    float* __restrict__ cs, int* __restrict__ canc, int* __restrict__ ccls) {
  int b = blockIdx.x;
  int tid = threadIdx.x;
  const float* base = scr + (size_t)b * ANCH_TOT * CCLS + (size_t)aoff * CCLS;
  __shared__ int scnt;
  if (tid == 0) scnt = 0;
  __syncthreads();
  int c0 = 0;
  for (int i = tid; i < lf; i += 256) if (base[i] > THRS) c0++;
  atomicAdd(&scnt, c0);
  __syncthreads();
  int cth = scnt;
  __syncthreads();
  float t = THRS;
  if (cth > kcap) {                       // binary search a score threshold
    float lo = THRS, hi = 1.0f;
    for (int it = 0; it < 30; ++it) {
      float mid = 0.5f * (lo + hi);
      if (tid == 0) scnt = 0;
      __syncthreads();
      int c = 0;
      for (int i = tid; i < lf; i += 256) if (base[i] > mid) c++;
      atomicAdd(&scnt, c);
      __syncthreads();
      int cc = scnt;
      __syncthreads();
      if (cc > kcap) lo = mid; else hi = mid;
    }
    t = hi;
  }
  // deterministic gather: per-thread count -> block scan -> ordered writes
  __shared__ int ssc[256];
  int mycnt = 0;
  for (int i = tid; i < lf; i += 256) if (base[i] > t) mycnt++;
  ssc[tid] = mycnt;
  __syncthreads();
  for (int off = 1; off < 256; off <<= 1) {
    int v = (tid >= off) ? ssc[tid - off] : 0;
    __syncthreads();
    ssc[tid] += v;
    __syncthreads();
  }
  int pos = ssc[tid] - mycnt;
  int tot = ssc[255];
  float* csB  = cs   + (size_t)b * KTOT + koff;
  int*  cancB = canc + (size_t)b * KTOT + koff;
  int*  cclsB = ccls + (size_t)b * KTOT + koff;
  for (int i = tid; i < lf; i += 256) {
    if (base[i] > t) {
      if (pos < kcap) {
        csB[pos]   = base[i];
        cancB[pos] = aoff + i / CCLS;
        cclsB[pos] = i % CCLS;
      }
      pos++;
    }
  }
  int start = tot < kcap ? tot : kcap;
  for (int o = start + tid; o < kcap; o += 256) { csB[o] = -1.0f; cancB[o] = 0; cclsB[o] = 0; }
}

// ----------------------------------------------------------------- decode
__global__ void decode_boxes(const float* __restrict__ cs, const int* __restrict__ canc,
                             const int* __restrict__ ccls, const float* __restrict__ priors,
                             const float* __restrict__ regf,
                             float* __restrict__ bx, float* __restrict__ boff,
                             float* __restrict__ sval) {
  int idx = blockIdx.x * 256 + threadIdx.x;
  if (idx >= BATCH * KTOT) return;
  int b = idx / KTOT;
  float v = cs[idx];
  int anc = canc[idx], c = ccls[idx];
  const float* pr = priors + (size_t)anc * 4;
  const float* rg = regf + ((size_t)b * ANCH_TOT + anc) * 4;
  float cx = pr[0] + rg[0] * 0.1f * pr[2];
  float cy = pr[1] + rg[1] * 0.1f * pr[3];
  float w = pr[2] * __expf(fminf(rg[2] * 0.2f, 4.135f));
  float h = pr[3] * __expf(fminf(rg[3] * 0.2f, 4.135f));
  float x1 = fminf(fmaxf(cx - 0.5f * w, 0.f), (float)IMG);
  float y1 = fminf(fmaxf(cy - 0.5f * h, 0.f), (float)IMG);
  float x2 = fminf(fmaxf(cx + 0.5f * w, 0.f), (float)IMG);
  float y2 = fminf(fmaxf(cy + 0.5f * h, 0.f), (float)IMG);
  bool valid = (v > THRS) && (x2 > x1) && (y2 > y1);
  float off = (float)c * (float)(IMG + 1);
  float* B = bx + (size_t)idx * 4;
  B[0] = x1; B[1] = y1; B[2] = x2; B[3] = y2;
  float* O = boff + (size_t)idx * 4;
  O[0] = x1 + off; O[1] = y1 + off; O[2] = x2 + off; O[3] = y2 + off;
  sval[idx] = valid ? v : -1.0f;
}

// ------------------------------------------------ bitonic sort (desc, stable)
__global__ __launch_bounds__(1024) void sort_desc(const float* __restrict__ sval,
                                                  float* __restrict__ skey, int* __restrict__ sidx) {
  int b = blockIdx.x;
  float* key = skey + (size_t)b * SORTN;
  int*  idx  = sidx + (size_t)b * SORTN;
  const float* sv = sval + (size_t)b * KTOT;
  for (int t = threadIdx.x; t < SORTN; t += 1024) {
    key[t] = (t < KTOT) ? sv[t] : -3.0e38f;
    idx[t] = t;
  }
  __syncthreads();
  for (int k = 2; k <= SORTN; k <<= 1) {
    for (int j = k >> 1; j > 0; j >>= 1) {
      for (int t = threadIdx.x; t < SORTN; t += 1024) {
        int p = t ^ j;
        if (p > t) {
          float ka = key[t], kb = key[p];
          int ia = idx[t], ib = idx[p];
          bool aLess = (ka < kb) || (ka == kb && ia > ib);
          bool desc = ((t & k) == 0);
          if (aLess == desc) { key[t] = kb; key[p] = ka; idx[t] = ib; idx[p] = ia; }
        }
      }
      __syncthreads();
    }
  }
}

// ----------------------------------------------------------------- NMS
__global__ __launch_bounds__(256) void nms_k(
    const float* __restrict__ skey, const int* __restrict__ sidx,
    const float* __restrict__ bx, const float* __restrict__ boff, const int* __restrict__ ccls,
    float* __restrict__ sbx, float* __restrict__ sboff, float* __restrict__ ss,
    int* __restrict__ scls, int* __restrict__ keep) {
  int b = blockIdx.x;
  const float* key = skey + (size_t)b * SORTN;
  const int*  sid  = sidx + (size_t)b * SORTN;
  const float* bxB = bx   + (size_t)b * KTOT * 4;
  const float* boB = boff + (size_t)b * KTOT * 4;
  const int*  clB  = ccls + (size_t)b * KTOT;
  float* sbxB = sbx   + (size_t)b * KTOT * 4;
  float* sboB = sboff + (size_t)b * KTOT * 4;
  float* ssB  = ss    + (size_t)b * KTOT;
  int*  sclB  = scls  + (size_t)b * KTOT;
  int*  kpB   = keep  + (size_t)b * KTOT;
  for (int j = threadIdx.x; j < KTOT; j += 256) {
    int p = sid[j];
    ssB[j]  = key[j];
    sclB[j] = clB[p];
    for (int q = 0; q < 4; ++q) { sbxB[j*4+q] = bxB[p*4+q]; sboB[j*4+q] = boB[p*4+q]; }
    kpB[j] = (key[j] > 0.0f) ? 1 : 0;
  }
  __syncthreads();
  for (int i = 0; i < KTOT; ++i) {
    if (kpB[i]) {
      float ax1 = sboB[i*4], ay1 = sboB[i*4+1], ax2 = sboB[i*4+2], ay2 = sboB[i*4+3];
      float aar = (ax2 - ax1) * (ay2 - ay1);
      for (int j = i + 1 + threadIdx.x; j < KTOT; j += 256) {
        if (!kpB[j]) continue;
        float bx1 = sboB[j*4], by1 = sboB[j*4+1], bx2 = sboB[j*4+2], by2 = sboB[j*4+3];
        float ix = fminf(ax2, bx2) - fmaxf(ax1, bx1);
        float iy = fminf(ay2, by2) - fmaxf(ay1, by1);
        float inter = fmaxf(ix, 0.f) * fmaxf(iy, 0.f);
        float bar = (bx2 - bx1) * (by2 - by1);
        float iou = inter / fmaxf(aar + bar - inter, 1e-12f);
        if (iou > IOU_THR) kpB[j] = 0;
      }
    }
    __syncthreads();
  }
}

// ----------------------------------------------------------------- finalize
__global__ void finalize_k(const float* __restrict__ sbx, const float* __restrict__ ss,
                           const int* __restrict__ scls, const int* __restrict__ keep,
                           float* __restrict__ oBox, float* __restrict__ oScr, int* __restrict__ oCls) {
  int b = blockIdx.x;
  if (threadIdx.x != 0) return;
  const float* sbxB = sbx  + (size_t)b * KTOT * 4;
  const float* ssB  = ss   + (size_t)b * KTOT;
  const int*  sclB  = scls + (size_t)b * KTOT;
  const int*  kpB   = keep + (size_t)b * KTOT;
  int outn = 0;
  for (int j = 0; j < KTOT && outn < MAXDET; ++j) {
    if (kpB[j] && ssB[j] > 0.0f) {
      for (int q = 0; q < 4; ++q) oBox[((size_t)b * MAXDET + outn) * 4 + q] = sbxB[j*4+q];
      oScr[(size_t)b * MAXDET + outn] = ssB[j];
      oCls[(size_t)b * MAXDET + outn] = sclB[j];
      outn++;
    }
  }
  for (; outn < MAXDET; ++outn) {
    for (int q = 0; q < 4; ++q) oBox[((size_t)b * MAXDET + outn) * 4 + q] = 0.f;
    oScr[(size_t)b * MAXDET + outn] = 0.f;
    oCls[(size_t)b * MAXDET + outn] = -1;
  }
}

// ================================================================== host
extern "C" void kernel_launch(void* const* d_in, const int* in_sizes, int n_in,
                              void* d_out, int out_size, void* d_ws, size_t ws_size,
                              hipStream_t stream) {
  (void)n_in; (void)out_size; (void)ws_size; (void)in_sizes;
  char* ws = (char*)d_ws;
  size_t cur = 0;
  auto alloc = [&](size_t bytes) -> size_t {
    size_t o = (cur + 255) & ~(size_t)255;
    cur = o + bytes;
    return o;
  };
  auto U = [&](size_t off) { return (ushort_t*)(ws + off); };
  auto F = [&](size_t off) { return (float*)(ws + off); };
  auto I = [&](size_t off) { return (int*)(ws + off); };

  // ---- conv descriptors; d_in layout: imgs, 15x(vgg w,b), l2_scale,
  //      4x(extra w1,b1,w2,b2), 6x(cls w,b), 6x(reg w,b)
  struct CD { int wi, bi, cin, cout, k; };
  CD cd[35];
  int nc = 0;
  const int vc[16] = {3,64,64,128,128,256,256,256,512,512,512,512,512,512,1024,1024};
  for (int i = 0; i < 15; ++i) cd[nc++] = {1 + 2*i, 2 + 2*i, vc[i], vc[i+1], (i == 14) ? 1 : 3};
  const int eci[4] = {1024, 512, 256, 256};
  const int eco[4] = {512, 256, 256, 256};
  for (int e = 0; e < 4; ++e) {
    cd[nc++] = {32 + 4*e, 33 + 4*e, eci[e], eco[e] / 2, 1};
    cd[nc++] = {34 + 4*e, 35 + 4*e, eco[e] / 2, eco[e], 3};
  }
  const int couts[6] = {512, 1024, 512, 256, 256, 256};
  const int nanch[6] = {4, 6, 6, 6, 4, 4};
  for (int i = 0; i < 6; ++i) cd[nc++] = {48 + 2*i, 49 + 2*i, couts[i], nanch[i] * NCLS, 3};
  for (int i = 0; i < 6; ++i) cd[nc++] = {60 + 2*i, 61 + 2*i, couts[i], nanch[i] * 4, 3};

  // padded bf16 weight matrices [Mpad x Kpad]
  size_t offW[35];
  int kpadA[35], mpadA[35];
  for (int i = 0; i < 35; ++i) {
    int Kdim = cd[i].cin * cd[i].k * cd[i].k;
    int Kpad = (Kdim + 31) & ~31;
    int Mpad = (cd[i].cout + 127) & ~127;
    kpadA[i] = Kpad; mpadA[i] = Mpad;
    offW[i] = alloc((size_t)Mpad * Kpad * 2);
  }

  const size_t ACT = (size_t)BATCH * 64 * IMG * IMG;
  size_t offA   = alloc(ACT * 2);
  size_t offB   = alloc(ACT * 2);
  size_t offImg = alloc((size_t)BATCH * 3 * IMG * IMG * 2);
  size_t offF0  = alloc((size_t)BATCH * 512 * 38 * 38 * 2);
  size_t offX19 = alloc((size_t)BATCH * 1024 * 19 * 19 * 2);
  size_t offX10 = alloc((size_t)BATCH * 512 * 10 * 10 * 2);
  size_t offX5  = alloc((size_t)BATCH * 256 * 5 * 5 * 2);
  size_t offX3  = alloc((size_t)BATCH * 256 * 3 * 3 * 2);
  size_t offX1  = alloc((size_t)BATCH * 256 * 2);
  size_t offHT  = alloc((size_t)2000000 * 4);
  size_t offScr = alloc((size_t)BATCH * ANCH_TOT * CCLS * 4);
  size_t offReg = alloc((size_t)BATCH * ANCH_TOT * 4 * 4);
  size_t offPri = alloc((size_t)ANCH_TOT * 4 * 4);
  size_t offCS  = alloc((size_t)BATCH * KTOT * 4);
  size_t offCA  = alloc((size_t)BATCH * KTOT * 4);
  size_t offCC  = alloc((size_t)BATCH * KTOT * 4);
  size_t offBX  = alloc((size_t)BATCH * KTOT * 4 * 4);
  size_t offBO  = alloc((size_t)BATCH * KTOT * 4 * 4);
  size_t offSV  = alloc((size_t)BATCH * KTOT * 4);
  size_t offSK  = alloc((size_t)BATCH * SORTN * 4);
  size_t offSI  = alloc((size_t)BATCH * SORTN * 4);
  size_t offSBX = alloc((size_t)BATCH * KTOT * 4 * 4);
  size_t offSBO = alloc((size_t)BATCH * KTOT * 4 * 4);
  size_t offSS  = alloc((size_t)BATCH * KTOT * 4);
  size_t offSC  = alloc((size_t)BATCH * KTOT * 4);
  size_t offKP  = alloc((size_t)BATCH * KTOT * 4);

  // repack weights (fp32 -> padded bf16), image -> bf16
  for (int i = 0; i < 35; ++i) {
    int Kdim = cd[i].cin * cd[i].k * cd[i].k;
    int tot = mpadA[i] * kpadA[i];
    repack_w<<<(tot + 255) / 256, 256, 0, stream>>>((const float*)d_in[cd[i].wi], U(offW[i]),
                                                    Kdim, kpadA[i], cd[i].cout, mpadA[i]);
  }
  { int n = BATCH * 3 * IMG * IMG;
    cvt_f32_bf16<<<(n + 255) / 256, 256, 0, stream>>>((const float*)d_in[0], U(offImg), n); }

  auto conv = [&](int ci, size_t inOff, int Hin, int Win, int Hout, int Wout,
                  int s, int p, int d, int relu, size_t outBfOff, float* outF) {
    const CD& c = cd[ci];
    int Ntot = BATCH * Hout * Wout;
    dim3 g((unsigned)((Ntot + 63) / 64), (unsigned)(mpadA[ci] / 128), 1);
    conv_bf16_wmma<<<g, dim3(128), 0, stream>>>(
        U(inOff), U(offW[ci]), (const float*)d_in[c.bi],
        (outBfOff == (size_t)-1) ? (ushort_t*)nullptr : U(outBfOff), outF,
        c.cin, Hin, Win, c.cout, Hout, Wout, c.k, s, p, d, relu, kpadA[ci]);
  };
  auto pool = [&](size_t inOff, size_t outOff, int C_, int Hin, int Win,
                  int Hout, int Wout, int k, int s, int p) {
    int tot = BATCH * C_ * Hout * Wout;
    maxpool_bf16<<<(tot + 255) / 256, 256, 0, stream>>>(U(inOff), U(outOff), C_, Hin, Win, Hout, Wout, k, s, p);
  };

  // ---- VGG backbone
  conv(0, offImg, 300, 300, 300, 300, 1, 1, 1, 1, offA, nullptr);
  conv(1, offA,   300, 300, 300, 300, 1, 1, 1, 1, offB, nullptr);
  pool(offB, offA, 64, 300, 300, 150, 150, 2, 2, 0);
  conv(2, offA, 150, 150, 150, 150, 1, 1, 1, 1, offB, nullptr);
  conv(3, offB, 150, 150, 150, 150, 1, 1, 1, 1, offA, nullptr);
  pool(offA, offB, 128, 150, 150, 75, 75, 2, 2, 0);
  conv(4, offB, 75, 75, 75, 75, 1, 1, 1, 1, offA, nullptr);
  conv(5, offA, 75, 75, 75, 75, 1, 1, 1, 1, offB, nullptr);
  conv(6, offB, 75, 75, 75, 75, 1, 1, 1, 1, offA, nullptr);
  pool(offA, offB, 256, 75, 75, 38, 38, 2, 2, 0);      // ceil mode via OOB skip
  conv(7, offB, 38, 38, 38, 38, 1, 1, 1, 1, offA, nullptr);
  conv(8, offA, 38, 38, 38, 38, 1, 1, 1, 1, offB, nullptr);
  conv(9, offB, 38, 38, 38, 38, 1, 1, 1, 1, offA, nullptr);   // f0 raw
  { int tot = BATCH * 38 * 38;
    l2norm_bf16<<<(tot + 255) / 256, 256, 0, stream>>>(U(offA), (const float*)d_in[31], U(offF0), 512, 38 * 38); }
  pool(offA, offB, 512, 38, 38, 19, 19, 2, 2, 0);
  conv(10, offB, 19, 19, 19, 19, 1, 1, 1, 1, offA, nullptr);
  conv(11, offA, 19, 19, 19, 19, 1, 1, 1, 1, offB, nullptr);
  conv(12, offB, 19, 19, 19, 19, 1, 1, 1, 1, offA, nullptr);
  pool(offA, offB, 512, 19, 19, 19, 19, 3, 1, 1);
  conv(13, offB, 19, 19, 19, 19, 1, 6, 6, 1, offA, nullptr);  // dilation 6
  conv(14, offA, 19, 19, 19, 19, 1, 0, 1, 1, offX19, nullptr);
  // ---- extras
  conv(15, offX19, 19, 19, 19, 19, 1, 0, 1, 1, offA, nullptr);
  conv(16, offA,   19, 19, 10, 10, 2, 1, 1, 1, offX10, nullptr);
  conv(17, offX10, 10, 10, 10, 10, 1, 0, 1, 1, offA, nullptr);
  conv(18, offA,   10, 10, 5, 5,   2, 1, 1, 1, offX5, nullptr);
  conv(19, offX5,  5, 5, 5, 5,     1, 0, 1, 1, offA, nullptr);
  conv(20, offA,   5, 5, 3, 3,     1, 0, 1, 1, offX3, nullptr);
  conv(21, offX3,  3, 3, 3, 3,     1, 0, 1, 1, offA, nullptr);
  conv(22, offA,   3, 3, 1, 1,     1, 0, 1, 1, offX1, nullptr);

  // ---- heads
  const int feats[6] = {38, 19, 10, 5, 3, 1};
  const size_t xsOff[6] = {offF0, offX19, offX10, offX5, offX3, offX1};
  const int aoffs[6] = {0, 5776, 7942, 8542, 8692, 8728};
  for (int l = 0; l < 6; ++l) {
    int f = feats[l];
    conv(23 + l, xsOff[l], f, f, f, f, 1, 1, 1, 0, (size_t)-1, F(offHT));
    { int tot = BATCH * f * f * nanch[l];
      head_softmax<<<(tot + 255) / 256, 256, 0, stream>>>(F(offHT), F(offScr), nanch[l], f, f, aoffs[l]); }
    conv(29 + l, xsOff[l], f, f, f, f, 1, 1, 1, 0, (size_t)-1, F(offHT));
    { int tot = BATCH * f * f * nanch[l];
      head_reg<<<(tot + 255) / 256, 256, 0, stream>>>(F(offHT), F(offReg), nanch[l], f, f, aoffs[l]); }
  }

  // ---- priors
  const float strf[6] = {8, 16, 32, 64, 100, 300};
  const float bsv[6]  = {21, 45, 99, 153, 207, 261};
  const float bnv[6]  = {45, 99, 153, 207, 261, 315};
  for (int l = 0; l < 6; ++l) {
    int tot = feats[l] * feats[l] * nanch[l];
    build_priors<<<(tot + 255) / 256, 256, 0, stream>>>(F(offPri), aoffs[l], feats[l], nanch[l],
                                                        strf[l], bsv[l], bnv[l]);
  }

  // ---- per-level top-k
  const int lvlen[6] = {5776, 2166, 600, 150, 36, 4};
  const int kper[6]  = {1000, 1000, 1000, 1000, 1000, 320};
  int koff = 0;
  for (int l = 0; l < 6; ++l) {
    topk_level<<<BATCH, 256, 0, stream>>>(F(offScr), lvlen[l] * CCLS, kper[l], aoffs[l], koff,
                                          F(offCS), I(offCA), I(offCC));
    koff += kper[l];
  }

  // ---- decode / sort / NMS / finalize
  { int tot = BATCH * KTOT;
    decode_boxes<<<(tot + 255) / 256, 256, 0, stream>>>(F(offCS), I(offCA), I(offCC), F(offPri),
                                                        F(offReg), F(offBX), F(offBO), F(offSV)); }
  sort_desc<<<BATCH, 1024, 0, stream>>>(F(offSV), F(offSK), I(offSI));
  nms_k<<<BATCH, 256, 0, stream>>>(F(offSK), I(offSI), F(offBX), F(offBO), I(offCC),
                                   F(offSBX), F(offSBO), F(offSS), I(offSC), I(offKP));
  float* oBox = (float*)d_out;
  float* oScr = oBox + (size_t)BATCH * MAXDET * 4;
  int*  oCls  = (int*)(oScr + (size_t)BATCH * MAXDET);
  finalize_k<<<BATCH, 32, 0, stream>>>(F(offSBX), F(offSS), I(offSC), I(offKP), oBox, oScr, oCls);
}
